// TemporalGNN_51891794870979
// MI455X (gfx1250) — matrix operand order repaired and checked by reading, not compiled, not run on hardware
//
#include <hip/hip_runtime.h>
#include <hip/hip_bf16.h>

// ---------------------------------------------------------------------------
// TemporalGNN for MI455X (gfx1250): GCN(64->128)+BN+ReLU -> GCN(128->128)+BN
// -> mean pool -> 2-layer LSTM -> linear head.
// GEMMs use v_wmma_f32_16x16x32_f16 (f16 in, f32 acc) with weights staged in
// LDS (B fragments via ds_load_b128, A streamed via global_load_b128), and the
// GCN self-loop term fused into the GEMM epilogue. Graph aggregation is an
// atomic scatter. Everything launches on `stream`, graph-capture safe.
// ---------------------------------------------------------------------------

#define T_STEPS 8
#define NNODES 20000
#define NEDGES 320000
#define D_IN 64
#define H 128
#define BN_EPS 1e-5f

typedef _Float16 v16h __attribute__((ext_vector_type(16)));
typedef _Float16 v8h  __attribute__((ext_vector_type(8)));
typedef float    v8f  __attribute__((ext_vector_type(8)));

// -------------------------------- helpers ----------------------------------

__device__ __forceinline__ v16h load_frag16(const _Float16* __restrict__ p, int secondHalfOff) {
    // two contiguous 16-byte loads -> one 16-half WMMA fragment
    v8h lo = *(const v8h*)(p);
    v8h hi = *(const v8h*)(p + secondHalfOff);
    v16h r;
#pragma unroll
    for (int i = 0; i < 8; ++i) { r[i] = lo[i]; r[8 + i] = hi[i]; }
    return r;
}

__device__ __forceinline__ v16h load_frag16_lds(const _Float16* p, int secondHalfOff) {
    v8h lo = *(const v8h*)(p);
    v8h hi = *(const v8h*)(p + secondHalfOff);
    v16h r;
#pragma unroll
    for (int i = 0; i < 8; ++i) { r[i] = lo[i]; r[8 + i] = hi[i]; }
    return r;
}

__device__ __forceinline__ float sigmoidf_(float x) { return 1.0f / (1.0f + expf(-x)); }

// ----------------------------- pointwise prep ------------------------------

__global__ void k_f32_to_f16(const float* __restrict__ in, _Float16* __restrict__ out, int n) {
    int i = blockIdx.x * blockDim.x + threadIdx.x;
    if (i < n) out[i] = (_Float16)in[i];
}

// W is [K x 128] row-major (f32). Produce Wt [128 x K] row-major f16 so that
// the WMMA B-fragment (per-column K-runs) is a contiguous load.
__global__ void k_transpose_f16(const float* __restrict__ W, _Float16* __restrict__ Wt, int K) {
    int i = blockIdx.x * blockDim.x + threadIdx.x;
    if (i >= K * H) return;
    int k = i / H, n = i % H;
    Wt[n * K + k] = (_Float16)W[i];
}

__global__ void k_zero_f32(float* __restrict__ p, int n) {
    int i = blockIdx.x * blockDim.x + threadIdx.x;
    if (i < n) p[i] = 0.0f;
}

// ------------------------------- degree ------------------------------------

__global__ void k_deg_init(float* __restrict__ deg, int n) {
    int i = blockIdx.x * blockDim.x + threadIdx.x;
    if (i < n) deg[i] = 1.0f;                      // self loop
}

__global__ void k_deg_edges(const int* __restrict__ dst, float* __restrict__ deg, int e) {
    int i = blockIdx.x * blockDim.x + threadIdx.x;
    if (i < e) atomicAdd(&deg[dst[i]], 1.0f);
}

__global__ void k_deg_to_dinv(float* __restrict__ deg, int n) {
    int i = blockIdx.x * blockDim.x + threadIdx.x;
    if (i < n) deg[i] = rsqrtf(deg[i]);            // in place: deg -> D^{-1/2}
}

// ------------------------------- WMMA GEMM ---------------------------------
// C[nrows x 128] = A16[nrows x K] * B  (Bt16 is [128 x K] = B transposed).
// Weights staged in LDS once per workgroup; B fragments read via ds_load_b128.
// Epilogue also writes agg[m][:] = C[m][:] * dinv[m]^2  (GCN self-loop term),
// which both initializes agg and saves a full extra pass over C.
// One wave computes a 16x128 slab: 8 f32 accumulator tiles, K in chunks of 32.
__global__ void __launch_bounds__(256)
k_gemm_wmma_f16(const _Float16* __restrict__ A16, const _Float16* __restrict__ Bt16,
                float* __restrict__ C, float* __restrict__ agg,
                const float* __restrict__ dinv, int nrows, int K) {
    __shared__ _Float16 Blds[H * H];               // up to 32 KB (K<=128)

    // cooperative stage of Bt16 (128*K halves) into LDS, 16B per thread-step
    {
        const int chunks = (H * K) / 8;            // # of 16-byte chunks
        const uint4* src = (const uint4*)Bt16;
        uint4* dstp = (uint4*)Blds;
        for (int i = threadIdx.x; i < chunks; i += 256) dstp[i] = src[i];
    }
    __syncthreads();

    const int lane = threadIdx.x & 31;
    const int wid  = threadIdx.x >> 5;
    const int r0   = (blockIdx.x * 8 + wid) * 16;
    if (r0 >= nrows) return;

    const int halfSel = (lane < 16) ? 0 : 1;       // which half-wave
    const int row     = r0 + (lane & 15);          // A: one row per lane
    const int colInT  = lane & 15;                 // B/C: one column per lane

    v8f acc[8];
#pragma unroll
    for (int j = 0; j < 8; ++j)
#pragma unroll
        for (int i = 0; i < 8; ++i) acc[j][i] = 0.0f;

    for (int kb = 0; kb < K; kb += 32) {
        // A fragment: lanes 0-15 hold K {0..7, 16..23}, lanes 16-31 {8..15, 24..31}
        const _Float16* pa = A16 + (size_t)row * K + kb + (halfSel ? 8 : 0);
        v16h a = load_frag16(pa, 16);
#pragma unroll
        for (int j = 0; j < 8; ++j) {
            // B fragment: lanes 0-15 hold K {0..7, 16..23}, lanes 16-31 {8..15, 24..31}
            const int n = j * 16 + colInT;
            const _Float16* pb = Blds + n * K + kb + (halfSel ? 8 : 0);
            v16h b = load_frag16_lds(pb, 16);
            acc[j] = __builtin_amdgcn_wmma_f32_16x16x32_f16(
                false, a, false, b, (short)0, acc[j], false, false);
        }
    }

    // D layout: VGPR r -> M = r (+8 for upper half-wave), lane -> N.
    // Fused epilogue: write dense result C and self-loop-scaled agg.
#pragma unroll
    for (int r = 0; r < 8; ++r) {
        const int m = r0 + r + (halfSel ? 8 : 0);
        float w = dinv[m]; w *= w;
        float* crow = C + (size_t)m * H + colInT;
        float* arow = agg + (size_t)m * H + colInT;
#pragma unroll
        for (int j = 0; j < 8; ++j) {
            float val = acc[j][r];
            crow[j * 16] = val;
            arow[j * 16] = val * w;
        }
    }
}

// ---------------------------- graph aggregation ----------------------------

// agg[dst][:] += hW[src][:] * dinv[src]*dinv[dst]   (32 threads / edge, float4)
__global__ void k_agg_edges(const int* __restrict__ src, const int* __restrict__ dst,
                            const float* __restrict__ dinv, const float* __restrict__ hW,
                            float* __restrict__ agg, int e) {
    int i = blockIdx.x * blockDim.x + threadIdx.x;
    if (i >= e * 32) return;
    int ed = i >> 5, c = i & 31;
    int s = src[ed], d = dst[ed];
    float coef = dinv[s] * dinv[d];
    float4 v = ((const float4*)(hW + (size_t)s * H))[c];
    float* out = agg + (size_t)d * H + c * 4;
    atomicAdd(out + 0, v.x * coef);
    atomicAdd(out + 1, v.y * coef);
    atomicAdd(out + 2, v.z * coef);
    atomicAdd(out + 3, v.w * coef);
}

// ------------------------- BN (+ReLU) -> f16 for next GEMM -----------------

__global__ void k_bn_relu_f16(const float* __restrict__ agg,
                              const float* __restrict__ b, const float* __restrict__ g,
                              const float* __restrict__ be, const float* __restrict__ m,
                              const float* __restrict__ v, _Float16* __restrict__ out, int total) {
    int i = blockIdx.x * blockDim.x + threadIdx.x;
    if (i >= total) return;
    int f = i & (H - 1);
    float val = (agg[i] + b[f] - m[f]) * rsqrtf(v[f] + BN_EPS) * g[f] + be[f];
    out[i] = (_Float16)fmaxf(val, 0.0f);
}

// BN2 then feature-wise partial sums (64 rows / block), atomicAdd into emb[t].
__global__ void k_bn_pool(const float* __restrict__ agg,
                          const float* __restrict__ b, const float* __restrict__ g,
                          const float* __restrict__ be, const float* __restrict__ m,
                          const float* __restrict__ v, float* __restrict__ emb_t, int n) {
    __shared__ float red[256];
    const int f = threadIdx.x & (H - 1);
    const int rep = threadIdx.x >> 7;                        // 0 or 1
    const float sc = rsqrtf(v[f] + BN_EPS) * g[f];
    const float sh = be[f] + (b[f] - m[f]) * sc;
    float local = 0.0f;
    for (int r = rep; r < 64; r += 2) {
        int node = blockIdx.x * 64 + r;
        if (node < n) local += agg[(size_t)node * H + f] * sc + sh;
    }
    red[threadIdx.x] = local;
    __syncthreads();
    if (rep == 0) atomicAdd(&emb_t[f], red[threadIdx.x] + red[threadIdx.x + 128]);
}

// ------------------------------ LSTM head ----------------------------------
// Single block, 128 threads: thread j owns hidden unit j. Gate order i,f,g,o.
__global__ void k_lstm_head(const float* __restrict__ emb,   // [T x H] (sums over nodes)
                            const float* __restrict__ Wih0, const float* __restrict__ Whh0,
                            const float* __restrict__ bih0, const float* __restrict__ bhh0,
                            const float* __restrict__ Wih1, const float* __restrict__ Whh1,
                            const float* __restrict__ bih1, const float* __restrict__ bhh1,
                            const float* __restrict__ Wc,   const float* __restrict__ bc,
                            float* __restrict__ out) {      // [2 + H]
    __shared__ float xs[T_STEPS][H];
    __shared__ float hbuf[H];
    const int j = threadIdx.x;
    const float invN = 1.0f / (float)NNODES;
    for (int t = 0; t < T_STEPS; ++t) xs[t][j] = emb[t * H + j] * invN;

    for (int layer = 0; layer < 2; ++layer) {
        const float* Wih = layer ? Wih1 : Wih0;
        const float* Whh = layer ? Whh1 : Whh0;
        const float* bih = layer ? bih1 : bih0;
        const float* bhh = layer ? bhh1 : bhh0;
        float c = 0.0f;
        hbuf[j] = 0.0f;
        __syncthreads();
        for (int t = 0; t < T_STEPS; ++t) {
            float gi = bih[0 * H + j] + bhh[0 * H + j];
            float gf = bih[1 * H + j] + bhh[1 * H + j];
            float gg = bih[2 * H + j] + bhh[2 * H + j];
            float go = bih[3 * H + j] + bhh[3 * H + j];
            const float* wi = Wih + (size_t)(0 * H + j) * H;
            const float* wf = Wih + (size_t)(1 * H + j) * H;
            const float* wg = Wih + (size_t)(2 * H + j) * H;
            const float* wo = Wih + (size_t)(3 * H + j) * H;
            const float* ui = Whh + (size_t)(0 * H + j) * H;
            const float* uf = Whh + (size_t)(1 * H + j) * H;
            const float* ug = Whh + (size_t)(2 * H + j) * H;
            const float* uo = Whh + (size_t)(3 * H + j) * H;
            for (int k = 0; k < H; ++k) {
                float xv = xs[t][k], hv = hbuf[k];
                gi += wi[k] * xv + ui[k] * hv;
                gf += wf[k] * xv + uf[k] * hv;
                gg += wg[k] * xv + ug[k] * hv;
                go += wo[k] * xv + uo[k] * hv;
            }
            c = sigmoidf_(gf) * c + sigmoidf_(gi) * tanhf(gg);
            float h = sigmoidf_(go) * tanhf(c);
            __syncthreads();
            hbuf[j] = h;
            xs[t][j] = h;                    // ys of layer 0 become x of layer 1
            __syncthreads();
        }
    }
    // hbuf now holds h_T of layer 1
    out[2 + j] = hbuf[j];
    __syncthreads();
    if (j < 2) {
        float o = bc[j];
        for (int k = 0; k < H; ++k) o += Wc[(size_t)j * H + k] * hbuf[k];
        out[j] = o;
    }
}

// ------------------------------- launcher ----------------------------------

static inline int cdiv(int a, int b) { return (a + b - 1) / b; }

extern "C" void kernel_launch(void* const* d_in, const int* in_sizes, int n_in,
                              void* d_out, int out_size, void* d_ws, size_t ws_size,
                              hipStream_t stream) {
    const float* x_seq = (const float*)d_in[0];
    const int*   eidx  = (const int*)d_in[1];
    const float* W1 = (const float*)d_in[2],  *b1 = (const float*)d_in[3];
    const float* g1 = (const float*)d_in[4],  *be1 = (const float*)d_in[5];
    const float* m1 = (const float*)d_in[6],  *v1 = (const float*)d_in[7];
    const float* W2 = (const float*)d_in[8],  *b2 = (const float*)d_in[9];
    const float* g2 = (const float*)d_in[10], *be2 = (const float*)d_in[11];
    const float* m2 = (const float*)d_in[12], *v2 = (const float*)d_in[13];
    const float* Wih0 = (const float*)d_in[14], *Whh0 = (const float*)d_in[15];
    const float* bih0 = (const float*)d_in[16], *bhh0 = (const float*)d_in[17];
    const float* Wih1 = (const float*)d_in[18], *Whh1 = (const float*)d_in[19];
    const float* bih1 = (const float*)d_in[20], *bhh1 = (const float*)d_in[21];
    const float* Wc = (const float*)d_in[22], *bc = (const float*)d_in[23];
    float* out = (float*)d_out;

    // ---- workspace bump allocation (256B aligned) ----
    char* ws = (char*)d_ws;
    size_t off = 0;
    auto take = [&](size_t bytes) -> char* {
        off = (off + 255) & ~(size_t)255;
        char* p = ws + off;
        off += bytes;
        return p;
    };
    _Float16* x16   = (_Float16*)take((size_t)T_STEPS * NNODES * D_IN * 2);  // 20.5 MB
    _Float16* W1t16 = (_Float16*)take((size_t)D_IN * H * 2);
    _Float16* W2t16 = (_Float16*)take((size_t)H * H * 2);
    float*    hW    = (float*)   take((size_t)NNODES * H * 4);               // 10.2 MB
    float*    agg   = (float*)   take((size_t)NNODES * H * 4);               // 10.2 MB
    _Float16* h1_16 = (_Float16*)take((size_t)NNODES * H * 2);               // 5.1 MB
    float*    dinv  = (float*)   take((size_t)NNODES * 4);
    float*    emb   = (float*)   take((size_t)T_STEPS * H * 4);
    (void)ws_size;

    const int NT = 256;

    // one-time prep (per launch; deterministic)
    {
        int n = T_STEPS * NNODES * D_IN;
        k_f32_to_f16<<<cdiv(n, NT), NT, 0, stream>>>(x_seq, x16, n);
        k_transpose_f16<<<cdiv(D_IN * H, NT), NT, 0, stream>>>(W1, W1t16, D_IN);
        k_transpose_f16<<<cdiv(H * H, NT), NT, 0, stream>>>(W2, W2t16, H);
        k_zero_f32<<<cdiv(T_STEPS * H, NT), NT, 0, stream>>>(emb, T_STEPS * H);
    }

    for (int t = 0; t < T_STEPS; ++t) {
        const int* src = eidx + (size_t)t * 2 * NEDGES;
        const int* dst = src + NEDGES;

        // degrees -> D^{-1/2}
        k_deg_init<<<cdiv(NNODES, NT), NT, 0, stream>>>(dinv, NNODES);
        k_deg_edges<<<cdiv(NEDGES, NT), NT, 0, stream>>>(dst, dinv, NEDGES);
        k_deg_to_dinv<<<cdiv(NNODES, NT), NT, 0, stream>>>(dinv, NNODES);

        // layer 1: GEMM (+fused self-loop init of agg) + scatter + BN + ReLU
        k_gemm_wmma_f16<<<cdiv(NNODES / 16, 8), NT, 0, stream>>>(
            x16 + (size_t)t * NNODES * D_IN, W1t16, hW, agg, dinv, NNODES, D_IN);
        k_agg_edges<<<cdiv(NEDGES * 32, NT), NT, 0, stream>>>(src, dst, dinv, hW, agg, NEDGES);
        k_bn_relu_f16<<<cdiv(NNODES * H, NT), NT, 0, stream>>>(
            agg, b1, g1, be1, m1, v1, h1_16, NNODES * H);

        // layer 2: GEMM (+fused self-loop init) + scatter + BN + mean pool
        k_gemm_wmma_f16<<<cdiv(NNODES / 16, 8), NT, 0, stream>>>(
            h1_16, W2t16, hW, agg, dinv, NNODES, H);
        k_agg_edges<<<cdiv(NEDGES * 32, NT), NT, 0, stream>>>(src, dst, dinv, hW, agg, NEDGES);
        k_bn_pool<<<cdiv(NNODES, 64), NT, 0, stream>>>(
            agg, b2, g2, be2, m2, v2, emb + (size_t)t * H, NNODES);
    }

    // sequential LSTM + classifier head; writes (out[2], hT[128]) = 130 floats
    k_lstm_head<<<1, H, 0, stream>>>(emb, Wih0, Whh0, bih0, bhh0,
                                     Wih1, Whh1, bih1, bhh1, Wc, bc, out);
    (void)in_sizes; (void)n_in; (void)out_size;
}
